// PAMS_EDSR_71597104824683
// MI455X (gfx1250) — 16-bit and 8-bit matrix operand order repaired and checked by reading, compiled, not run on hardware
//
#include <hip/hip_runtime.h>
#include <stdint.h>

typedef __attribute__((ext_vector_type(8)))  int      v8i;
typedef __attribute__((ext_vector_type(8)))  float    v8f;
typedef __attribute__((ext_vector_type(16))) _Float16 v16h;
typedef __attribute__((ext_vector_type(8)))  _Float16 v8h;

#define NIMG 16
#define HW   128
#define HH   130
#define SH   258
#define NCONV 33

static constexpr size_t SZ_H    = (size_t)NIMG*HW*HW*64*4;
static constexpr size_t SZ_S    = (size_t)NIMG*HH*HH*64;
static constexpr size_t SZ_SHUF = (size_t)NIMG*SH*SH*64*2;
static constexpr size_t SZ_WBIN = (size_t)NCONV*9*4*32*32;
static constexpr size_t SZ_ALPHA= (size_t)NCONV*64*4;
static constexpr size_t OFF_H    = 0;
static constexpr size_t OFF_R    = OFF_H + SZ_H;
static constexpr size_t OFF_SA   = OFF_R + SZ_H;
static constexpr size_t OFF_SB   = OFF_SA + SZ_S;
static constexpr size_t OFF_SHUF = OFF_SB + SZ_S;
static constexpr size_t OFF_WBIN = OFF_SHUF + SZ_SHUF;
static constexpr size_t OFF_ALPHA= OFF_WBIN + SZ_WBIN;
static constexpr size_t OFF_WUP  = OFF_ALPHA + SZ_ALPHA;

__device__ __forceinline__ uint32_t sgn4pack(float a, float b, float c, float d) {
  uint32_t r;
  r  = (uint32_t)(((a > 0.f) - (a < 0.f)) & 0xFF);
  r |= (uint32_t)(((b > 0.f) - (b < 0.f)) & 0xFF) << 8;
  r |= (uint32_t)(((c > 0.f) - (c < 0.f)) & 0xFF) << 16;
  r |= (uint32_t)(((d > 0.f) - (d < 0.f)) & 0xFF) << 24;
  return r;
}

__global__ void zero_kernel(uint4* __restrict__ p, size_t n) {
  size_t i = (size_t)blockIdx.x * blockDim.x + threadIdx.x;
  size_t stride = (size_t)gridDim.x * blockDim.x;
  uint4 z; z.x = 0; z.y = 0; z.z = 0; z.w = 0;
  for (; i < n; i += stride) p[i] = z;
}

__global__ void alpha_kernel(const float* __restrict__ wbody,
                             const float* __restrict__ wlast,
                             float* __restrict__ alpha) {
  int id = blockIdx.x * blockDim.x + threadIdx.x;
  if (id >= NCONV * 64) return;
  int c = id >> 6, o = id & 63;
  const float* w = (c < 32) ? (wbody + (size_t)(c*64 + o)*576)
                            : (wlast + (size_t)o*576);
  float s = 0.f;
  for (int i = 0; i < 576; ++i) s += fabsf(w[i]);
  alpha[id] = s * (1.0f / 576.0f);
}

__global__ void packwbin_kernel(const float* __restrict__ wbody,
                                const float* __restrict__ wlast,
                                int8_t* __restrict__ wbin) {
  int id = blockIdx.x * blockDim.x + threadIdx.x;
  if (id >= NCONV * 9 * 4 * 32) return;
  int lane = id & 31;
  int gm   = (id >> 5) & 3;
  int t    = (id >> 7) % 9;
  int c    = id / 1152;
  int m = lane & 15, hi = lane >> 4;
  int o = gm * 16 + m;
  uint32_t* dst = (uint32_t*)(wbin + (size_t)id * 32);
  #pragma unroll
  for (int v = 0; v < 8; ++v) {
    uint32_t d = 0;
    #pragma unroll
    for (int b = 0; b < 4; ++b) {
      int k = ((v >> 2) << 5) + (hi << 4) + ((v & 3) << 2) + b;
      float w = (c < 32) ? wbody[((size_t)(c*64 + o)*64 + k)*9 + t]
                         : wlast[((size_t)o*64 + k)*9 + t];
      int s = (w > 0.f) - (w < 0.f);
      d |= (uint32_t)(s & 0xFF) << (8 * b);
    }
    dst[v] = d;
  }
}

__global__ void packwup_kernel(const float* __restrict__ wup_in,
                               _Float16* __restrict__ wup_out) {
  int id = blockIdx.x * blockDim.x + threadIdx.x;
  if (id >= 9 * 2 * 16 * 32) return;
  int lane = id & 31;
  int gm   = (id >> 5) & 15;
  int kt   = (id >> 9) & 1;
  int t    = id >> 10;
  int m = lane & 15, hi = lane >> 4;
  int o = gm * 16 + m;
  _Float16* dst = wup_out + (size_t)id * 16;
  #pragma unroll
  for (int hidx = 0; hidx < 16; ++hidx) {
    int v = hidx >> 1, hh = hidx & 1;
    int k = ((v >= 4) ? 16 : 0) + (hi << 3) + ((v & 3) << 1) + hh;
    int i = kt * 32 + k;
    dst[hidx] = (_Float16)wup_in[((size_t)o*64 + i)*9 + t];
  }
}

__global__ __launch_bounds__(256) void head_kernel(const float* __restrict__ x,
                                                   const float* __restrict__ wh,
                                                   float* __restrict__ hbuf,
                                                   float* __restrict__ rbuf,
                                                   int8_t* __restrict__ sA) {
  __shared__ float xs[4][27];
  int sub = threadIdx.x >> 6;
  int o   = threadIdx.x & 63;
  int pos = blockIdx.x * 4 + sub;
  int xc = pos & 127, yr = (pos >> 7) & 127, n = pos >> 14;
  if (o < 27) {
    int ci = o / 9, rem = o % 9, dy = rem / 3, dx = rem % 3;
    int yy = yr + dy - 1, xx = xc + dx - 1;
    float v = 0.f;
    if (yy >= 0 && yy < 128 && xx >= 0 && xx < 128) {
      float sc = (ci == 0) ? 114.444f : ((ci == 1) ? 111.4605f : 103.02f);
      v = x[(((size_t)n*3 + ci)*128 + yy)*128 + xx] - sc;
    }
    xs[sub][o] = v;
  }
  __syncthreads();
  float acc = 0.f;
  #pragma unroll
  for (int t = 0; t < 27; ++t) acc += xs[sub][t] * wh[o*27 + t];
  size_t idx = (size_t)pos * 64 + o;
  hbuf[idx] = acc;
  rbuf[idx] = acc;
  sA[(((size_t)n*HH + yr + 1)*HH + (xc + 1))*64 + o] =
      (int8_t)((acc > 0.f) - (acc < 0.f));
}

__global__ __launch_bounds__(256) void binconv_kernel(
    const int8_t* __restrict__ sIn, const int8_t* __restrict__ wbin, int conv,
    const float* __restrict__ alpha, float* __restrict__ rbuf,
    const float* __restrict__ hbuf, int8_t* __restrict__ sOut,
    float* __restrict__ resNCHW, int mode) {
  __shared__ int ldsw[9 * 4 * 32 * 8];
  const int tid = threadIdx.x;

  {
    uint64_t ga = (uint64_t)(uintptr_t)(wbin + (size_t)conv * 36864) +
                  (uint64_t)tid * 16;
    uint32_t la = (uint32_t)(uintptr_t)(&ldsw[0]) + (uint32_t)tid * 16;
    #pragma unroll
    for (int i = 0; i < 9; ++i) {
      asm volatile("global_load_async_to_lds_b128 %0, %1, off"
                   :: "v"(la), "v"(ga) : "memory");
      ga += 4096;
      la += 4096;
    }
    if (conv < 32)
      __builtin_prefetch(wbin + (size_t)(conv + 1) * 36864 + (size_t)tid * 144, 0, 1);
    asm volatile("s_wait_asynccnt 0x0" ::: "memory");
  }
  __syncthreads();

  const int lane = tid & 31, wave = tid >> 5;
  const int tw = blockIdx.x * 8 + wave;
  const int x0 = (tw & 7) << 4;
  const int yr = (tw >> 3) & 127;
  const int n  = tw >> 10;
  const int m = lane & 15, hi = lane >> 4;

  v8i z = {0,0,0,0,0,0,0,0};
  v8i acc[4] = {z, z, z, z};

  #pragma unroll
  for (int t = 0; t < 9; ++t) {
    const int dy = t/3 - 1, dx = t%3 - 1;
    const int8_t* p = sIn +
      (((size_t)n*HH + (yr + 1 + dy))*HH + (x0 + m + 1 + dx))*64 + hi*16;
    int4 b0 = *(const int4*)p;
    int4 b1 = *(const int4*)(p + 32);
    v8i B = {b0.x, b0.y, b0.z, b0.w, b1.x, b1.y, b1.z, b1.w};
    #pragma unroll
    for (int g = 0; g < 4; ++g) {
      v8i A = *(const v8i*)&ldsw[((t*4 + g)*32 + lane)*8];
      acc[g] = __builtin_amdgcn_wmma_i32_16x16x64_iu8(true, A, true, B, acc[g],
                                                      false, false);
    }
  }

  const size_t posN = (((size_t)n*HW + yr)*HW + (x0 + m)) * 64;
  const size_t posH = (((size_t)n*HH + (yr + 1))*HH + (x0 + m + 1)) * 64;

  if (mode == 0) {
    #pragma unroll
    for (int g = 0; g < 4; ++g) {
      uint32_t lo = 0, hb = 0;
      #pragma unroll
      for (int v = 0; v < 4; ++v) {
        int s0 = (acc[g][v]   > 0) - (acc[g][v]   < 0);
        int s1 = (acc[g][v+4] > 0) - (acc[g][v+4] < 0);
        lo |= (uint32_t)(s0 & 0xFF) << (8*v);
        hb |= (uint32_t)(s1 & 0xFF) << (8*v);
      }
      uint2 u; u.x = lo; u.y = hb;
      *(uint2*)(sOut + posH + g*16 + hi*8) = u;
    }
  } else if (mode == 1) {
    const float* al = alpha + conv * 64;
    #pragma unroll
    for (int g = 0; g < 4; ++g) {
      const int o0 = g*16 + hi*8;
      float4 r0 = *(const float4*)(rbuf + posN + o0);
      float4 r1 = *(const float4*)(rbuf + posN + o0 + 4);
      float4 a0 = *(const float4*)(al + o0);
      float4 a1 = *(const float4*)(al + o0 + 4);
      r0.x += a0.x * (float)acc[g][0];  r0.y += a0.y * (float)acc[g][1];
      r0.z += a0.z * (float)acc[g][2];  r0.w += a0.w * (float)acc[g][3];
      r1.x += a1.x * (float)acc[g][4];  r1.y += a1.y * (float)acc[g][5];
      r1.z += a1.z * (float)acc[g][6];  r1.w += a1.w * (float)acc[g][7];
      *(float4*)(rbuf + posN + o0)     = r0;
      *(float4*)(rbuf + posN + o0 + 4) = r1;
      uint2 u;
      u.x = sgn4pack(r0.x, r0.y, r0.z, r0.w);
      u.y = sgn4pack(r1.x, r1.y, r1.z, r1.w);
      *(uint2*)(sOut + posH + o0) = u;
    }
  } else {
    const float* al = alpha + conv * 64;
    #pragma unroll
    for (int g = 0; g < 4; ++g) {
      const int o0 = g*16 + hi*8;
      float4 h0 = *(const float4*)(hbuf + posN + o0);
      float4 h1 = *(const float4*)(hbuf + posN + o0 + 4);
      float4 a0 = *(const float4*)(al + o0);
      float4 a1 = *(const float4*)(al + o0 + 4);
      h0.x += a0.x * (float)acc[g][0];  h0.y += a0.y * (float)acc[g][1];
      h0.z += a0.z * (float)acc[g][2];  h0.w += a0.w * (float)acc[g][3];
      h1.x += a1.x * (float)acc[g][4];  h1.y += a1.y * (float)acc[g][5];
      h1.z += a1.z * (float)acc[g][6];  h1.w += a1.w * (float)acc[g][7];
      *(float4*)(rbuf + posN + o0)     = h0;
      *(float4*)(rbuf + posN + o0 + 4) = h1;
      float rv[8] = {h0.x, h0.y, h0.z, h0.w, h1.x, h1.y, h1.z, h1.w};
      #pragma unroll
      for (int v = 0; v < 8; ++v)
        resNCHW[(((size_t)n*64 + o0 + v)*HW + yr)*HW + (x0 + m)] = rv[v];
    }
  }
}

__global__ __launch_bounds__(256) void upconv_kernel(
    const float* __restrict__ res, const _Float16* __restrict__ wup,
    _Float16* __restrict__ shuf) {
  const int tid = threadIdx.x;
  const int lane = tid & 31, wave = tid >> 5;
  const int gchunk = blockIdx.y;
  const int tw = blockIdx.x * 8 + wave;
  const int x0 = (tw & 7) << 4;
  const int yr = (tw >> 3) & 127;
  const int n  = tw >> 10;
  const int m = lane & 15, hi = lane >> 4;

  v8f z = {0,0,0,0,0,0,0,0};
  v8f acc[4] = {z, z, z, z};
  float4 fz; fz.x = 0.f; fz.y = 0.f; fz.z = 0.f; fz.w = 0.f;

  #pragma unroll
  for (int t = 0; t < 9; ++t) {
    const int dy = t/3 - 1, dx = t%3 - 1;
    const int yv = yr + dy, xv = x0 + m + dx;
    const bool valid = (yv >= 0) && (yv < 128) && (xv >= 0) && (xv < 128);
    const int yc = valid ? yv : 0, xc = valid ? xv : 0;
    const float4* bp = (const float4*)(res +
        (((size_t)n*HW + yc)*HW + xc)*64 + hi*8);
    #pragma unroll
    for (int kt = 0; kt < 2; ++kt) {
      float4 f0 = valid ? bp[kt*8 + 0] : fz;
      float4 f1 = valid ? bp[kt*8 + 1] : fz;
      float4 f2 = valid ? bp[kt*8 + 4] : fz;
      float4 f3 = valid ? bp[kt*8 + 5] : fz;
      v16h B;
      B[0]=(_Float16)f0.x;  B[1]=(_Float16)f0.y;  B[2]=(_Float16)f0.z;  B[3]=(_Float16)f0.w;
      B[4]=(_Float16)f1.x;  B[5]=(_Float16)f1.y;  B[6]=(_Float16)f1.z;  B[7]=(_Float16)f1.w;
      B[8]=(_Float16)f2.x;  B[9]=(_Float16)f2.y;  B[10]=(_Float16)f2.z; B[11]=(_Float16)f2.w;
      B[12]=(_Float16)f3.x; B[13]=(_Float16)f3.y; B[14]=(_Float16)f3.z; B[15]=(_Float16)f3.w;
      #pragma unroll
      for (int gl = 0; gl < 4; ++gl) {
        const v16h A = *(const v16h*)(
            wup + ((size_t)((t*2 + kt)*16 + gchunk*4 + gl)*32 + lane)*16);
        acc[gl] = __builtin_amdgcn_wmma_f32_16x16x32_f16(
            false, A, false, B, (short)0, acc[gl], false, false);
      }
    }
  }
  const int xcol = x0 + m;
  #pragma unroll
  for (int gl = 0; gl < 4; ++gl) {
    #pragma unroll
    for (int v = 0; v < 8; ++v) {
      int o  = gchunk*64 + gl*16 + hi*8 + v;
      int cp = o >> 2, ry = (o >> 1) & 1, rx = o & 1;
      size_t idx = (((size_t)n*SH + (2*yr + ry + 1))*SH + (2*xcol + rx + 1))*64 + cp;
      shuf[idx] = (_Float16)acc[gl][v];
    }
  }
}

__global__ __launch_bounds__(256) void tail_kernel(
    const _Float16* __restrict__ shuf, const float* __restrict__ wt,
    const float* __restrict__ bt, float* __restrict__ out) {
  int id = blockIdx.x * blockDim.x + threadIdx.x;
  int xo = id & 255, yo = (id >> 8) & 255, n = id >> 16;
  float a0 = 0.f, a1 = 0.f, a2 = 0.f;
  for (int t = 0; t < 9; ++t) {
    int dy = t / 3, dx = t % 3;
    const v8h* p = (const v8h*)(shuf +
        (((size_t)n*SH + (yo + dy))*SH + (xo + dx))*64);
    #pragma unroll
    for (int cb = 0; cb < 8; ++cb) {
      v8h hv = p[cb];
      #pragma unroll
      for (int j = 0; j < 8; ++j) {
        int c = cb*8 + j;
        float v = (float)hv[j];
        a0 += v * wt[(0*64 + c)*9 + t];
        a1 += v * wt[(1*64 + c)*9 + t];
        a2 += v * wt[(2*64 + c)*9 + t];
      }
    }
  }
  size_t base = (size_t)n * 3 * 65536 + (size_t)yo * 256 + xo;
  out[base]          = a0 + bt[0] + 114.444f;
  out[base + 65536]  = a1 + bt[1] + 111.4605f;
  out[base + 131072] = a2 + bt[2] + 103.02f;
}

extern "C" void kernel_launch(void* const* d_in, const int* in_sizes, int n_in,
                              void* d_out, int out_size, void* d_ws, size_t ws_size,
                              hipStream_t stream) {
  (void)in_sizes; (void)n_in; (void)out_size; (void)ws_size;
  const float* x     = (const float*)d_in[0];
  const float* whead = (const float*)d_in[1];
  const float* wbody = (const float*)d_in[2];
  const float* wlast = (const float*)d_in[3];
  const float* wupin = (const float*)d_in[4];
  const float* wtail = (const float*)d_in[5];
  const float* btail = (const float*)d_in[6];

  uint8_t* ws = (uint8_t*)d_ws;
  float*    hbuf  = (float*)(ws + OFF_H);
  float*    rbuf  = (float*)(ws + OFF_R);
  int8_t*   sA    = (int8_t*)(ws + OFF_SA);
  int8_t*   sB    = (int8_t*)(ws + OFF_SB);
  _Float16* shuf  = (_Float16*)(ws + OFF_SHUF);
  int8_t*   wbin  = (int8_t*)(ws + OFF_WBIN);
  float*    alpha = (float*)(ws + OFF_ALPHA);
  _Float16* wup   = (_Float16*)(ws + OFF_WUP);

  float* out_img = (float*)d_out;
  float* res_out = out_img + (size_t)16 * 3 * 256 * 256;

  zero_kernel<<<2048, 256, 0, stream>>>((uint4*)sA, SZ_S / 16);
  zero_kernel<<<2048, 256, 0, stream>>>((uint4*)sB, SZ_S / 16);
  zero_kernel<<<4096, 256, 0, stream>>>((uint4*)shuf, SZ_SHUF / 16);

  alpha_kernel  <<<(NCONV*64 + 255)/256, 256, 0, stream>>>(wbody, wlast, alpha);
  packwbin_kernel<<<(NCONV*1152 + 255)/256, 256, 0, stream>>>(wbody, wlast, wbin);
  packwup_kernel <<<(9*2*16*32 + 255)/256, 256, 0, stream>>>(wupin, wup);

  head_kernel<<<65536, 256, 0, stream>>>(x, whead, hbuf, rbuf, sA);

  for (int b = 0; b < 16; ++b) {
    binconv_kernel<<<2048, 256, 0, stream>>>(sA, wbin, 2*b,   alpha, rbuf, hbuf,
                                             sB, nullptr, 0);
    binconv_kernel<<<2048, 256, 0, stream>>>(sB, wbin, 2*b+1, alpha, rbuf, hbuf,
                                             sA, nullptr, 1);
  }
  binconv_kernel<<<2048, 256, 0, stream>>>(sA, wbin, 32, alpha, rbuf, hbuf,
                                           sB, res_out, 2);

  upconv_kernel<<<dim3(2048, 4), 256, 0, stream>>>(rbuf, wup, shuf);
  tail_kernel<<<4096, 256, 0, stream>>>(shuf, wtail, btail, out_img);
}
